// DenseDCRNNThenDiffConvModel_14396730376456
// MI455X (gfx1250) — matrix operand order repaired and checked by reading, not compiled, not run on hardware
//
#include <hip/hip_runtime.h>
#include <hip/hip_bf16.h>

// ---------------- model constants ----------------
#define BB 16
#define TT 12
#define NN 1024
#define FF 8
#define HH 64
#define EE (32 * NN)
#define HORIZON 12
#define COLS (BB * 2 * HH)   // 2048 columns in the diffusion GEMM

typedef __bf16 bf16_t;
typedef __attribute__((ext_vector_type(16))) __bf16 v16bf;
typedef __attribute__((ext_vector_type(8)))  float  v8f;
typedef __attribute__((ext_vector_type(4)))  unsigned uint32x4;
typedef __attribute__((ext_vector_type(8)))  int      int32x8;
typedef __attribute__((ext_vector_type(4)))  int      int32x4;

// =====================================================================
// Adjacency normalization
// =====================================================================
__global__ __launch_bounds__(256) void rowsum_kernel(const float* __restrict__ adj,
                                                     float* __restrict__ rsf,
                                                     float* __restrict__ rsb) {
  __shared__ float sf[256], sb[256];
  const int w = blockIdx.x;
  float af = 0.f, ab = 0.f;
  for (int v = threadIdx.x; v < NN; v += 256) {
    af += adj[w * NN + v];
    ab += adj[v * NN + w];
  }
  sf[threadIdx.x] = af; sb[threadIdx.x] = ab;
  __syncthreads();
  for (int s = 128; s > 0; s >>= 1) {
    if (threadIdx.x < s) { sf[threadIdx.x] += sf[threadIdx.x + s]; sb[threadIdx.x] += sb[threadIdx.x + s]; }
    __syncthreads();
  }
  if (threadIdx.x == 0) { rsf[w] = sf[0]; rsb[w] = sb[0]; }
}

__global__ __launch_bounds__(256) void norm_adj_kernel(const float* __restrict__ adj,
                                                       const float* __restrict__ rsf,
                                                       const float* __restrict__ rsb,
                                                       bf16_t* __restrict__ Af,
                                                       bf16_t* __restrict__ Ab) {
  const int idx = blockIdx.x * 256 + threadIdx.x;
  const int w = idx >> 10, v = idx & (NN - 1);
  Af[idx] = (bf16_t)(adj[idx] / rsf[w]);
  Ab[idx] = (bf16_t)(adj[v * NN + w] / rsb[w]);
}

// =====================================================================
// Weight transpose: W (K x 64, f32 row-major) -> Wt (64 x K, bf16 col-major-in-K)
// =====================================================================
__global__ __launch_bounds__(256) void transpose_w_kernel(const float* __restrict__ W,
                                                          bf16_t* __restrict__ Wt, int K) {
  const int idx = blockIdx.x * 256 + threadIdx.x;
  if (idx < K * 64) {
    const int c = idx >> 6, h = idx & 63;
    Wt[h * K + c] = (bf16_t)W[c * 64 + h];
  }
}

// =====================================================================
// Hop GEMM: Y = A(1024x1024 row-major bf16) @ X(1024 x 2048 col-major bf16)
// B tile (64 cols x 32 k = 4KB) DMA'd to LDS by the Tensor Data Mover,
// double-buffered with s_wait_tensorcnt; A fragments stay in registers.
// Block = 8 waves = 128 rows x 64 cols.
// =====================================================================
__device__ __forceinline__ void tdm_load_tile(unsigned lds_off, unsigned long long ga) {
  uint32x4 g0;
  g0[0] = 1u;                                   // count = 1 valid descriptor
  g0[1] = lds_off;                              // lds_addr
  g0[2] = (unsigned)ga;                         // global_addr[31:0]
  g0[3] = (unsigned)(ga >> 32) | 0x80000000u;   // global_addr[56:32] | type=2
  int32x8 g1;
  g1[0] = (int)0x00010000u;                     // data_size = 2 bytes
  g1[1] = (int)(1024u << 16);                   // tensor_dim0 = 1024 (low16 in [63:48])
  g1[2] = (int)(2048u << 16);                   // tensor_dim1 = 2048 (low16 in [95:80])
  g1[3] = (int)(32u << 16);                     // tile_dim0 = 32   ([127:112])
  g1[4] = 64;                                   // tile_dim1 = 64   ([143:128])
  g1[5] = 1024;                                 // tensor_dim0_stride = 1024
  g1[6] = 0;
  g1[7] = 0;
  int32x4 z4 = {0, 0, 0, 0};
  int32x8 z8 = {0, 0, 0, 0, 0, 0, 0, 0};
  __builtin_amdgcn_tensor_load_to_lds(g0, g1, z4, z4, z8, 0);
}

__global__ __launch_bounds__(256) void gemm_hop_kernel(const bf16_t* __restrict__ A,
                                                       const bf16_t* __restrict__ X,
                                                       bf16_t* __restrict__ Y) {
  __shared__ __align__(16) bf16_t sB[2][64 * 32];   // [buf][col_local*32 + k_local]
  const int tid = threadIdx.x;
  const int lane = tid & 31;
  const int wave = tid >> 5;
  const int rowBase = blockIdx.y * 128 + wave * 16;
  const int colBase = blockIdx.x * 64;
  const int g  = lane >> 4;
  const int ml = lane & 15;

  const unsigned lds0 = (unsigned)(unsigned long long)(void*)&sB[0][0];
  const unsigned long long xbase =
      (unsigned long long)(const void*)(X + (long)colBase * NN);

  // prologue: DMA tile k0=0 into buffer 0
  if (tid == 0) {
    tdm_load_tile(lds0, xbase);
    __builtin_amdgcn_s_wait_tensorcnt(0);
  }
  __syncthreads();

  v8f acc[4];
  acc[0] = (v8f){}; acc[1] = (v8f){}; acc[2] = (v8f){}; acc[3] = (v8f){};
  const unsigned* arow = (const unsigned*)(A + (long)(rowBase + ml) * NN);

  for (int k0 = 0; k0 < NN; k0 += 32) {
    const int cur = (k0 >> 5) & 1;
    if (tid == 0 && k0 + 32 < NN)                 // prefetch next tile (other buffer)
      tdm_load_tile(lds0 + (unsigned)((cur ^ 1) * 4096), xbase + (k0 + 32) * 2ull);

    // A fragment: row rowBase+ml; k = g*16 + v*2
    union { unsigned u[8]; v16bf v; } af;
    const unsigned* ak = arow + (k0 >> 1);
#pragma unroll
    for (int v2 = 0; v2 < 8; ++v2)
      af.u[v2] = ak[g * 8 + v2];

    // B fragments from LDS: col colBase+nt*16+ml, k = g*16..+15 contiguous
#pragma unroll
    for (int nt = 0; nt < 4; ++nt) {
      v16bf bf = *(const v16bf*)(&sB[cur][(nt * 16 + ml) * 32 + g * 16]);
      acc[nt] = __builtin_amdgcn_wmma_f32_16x16x32_bf16(
          false, af.v, false, bf, (short)0, acc[nt], false, false);
    }

    if (tid == 0 && k0 + 32 < NN)
      __builtin_amdgcn_s_wait_tensorcnt(0);       // next tile landed
    __syncthreads();                              // all waves done with cur, next visible
  }

#pragma unroll
  for (int nt = 0; nt < 4; ++nt) {
    const int col = colBase + nt * 16 + ml;
    bf16_t tmp[8];
#pragma unroll
    for (int q = 0; q < 8; ++q) tmp[q] = (bf16_t)acc[nt][q];
    *(uint4*)(Y + (long)col * NN + rowBase + g * 8) = *(const uint4*)tmp;
  }
}

// =====================================================================
// pack_xh: xt = x[:,t] @ W_enc + b_enc + node_emb;  X0 = concat(xt, h) col-major bf16
// =====================================================================
__global__ __launch_bounds__(256) void pack_xh_kernel(const float* __restrict__ x,
                                                      const float* __restrict__ Wenc,
                                                      const float* __restrict__ benc,
                                                      const float* __restrict__ nemb,
                                                      const float* __restrict__ hst,
                                                      bf16_t* __restrict__ X0, int t) {
  const int idx = blockIdx.x * 256 + threadIdx.x;  // (b*NN+n)*HH + h
  const int h = idx & 63;
  const int n = (idx >> 6) & (NN - 1);
  const int b = idx >> 16;
  const float* xp = x + ((((long)b * TT + t) * NN + n) * FF);
  float acc = benc[h] + nemb[n * HH + h];
#pragma unroll
  for (int f = 0; f < FF; ++f) acc += xp[f] * Wenc[f * HH + h];
  X0[((long)(b * 128 + h)) * NN + n]      = (bf16_t)acc;
  X0[((long)(b * 128 + 64 + h)) * NN + n] = (bf16_t)hst[idx];
}

// =====================================================================
// Feats tile loader (shared by gate kernels): 64 bn-rows x 512 channels -> LDS
// feat[(b,n), c] = Yhop[(b*128 + (c&127))*NN + n]; contiguous along n.
// =====================================================================
__device__ __forceinline__ void load_feats_tile(bf16_t* sF,
                                                const bf16_t* Y1f, const bf16_t* Y2f,
                                                const bf16_t* Y1b, const bf16_t* Y2b,
                                                int b, int n0, int tid) {
  const bf16_t* Ys[4] = {Y1f, Y2f, Y1b, Y2b};
  for (int e = tid; e < 16384; e += 256) {       // uint elements: 2 rows each
    const int rp = e & 31;                        // row pair
    const int c  = e >> 5;
    const unsigned v = *(const unsigned*)(Ys[c >> 7] +
        ((long)(b * 128 + (c & 127))) * NN + n0 + rp * 2);
    sF[(rp * 2) * 512 + c]     = ((const bf16_t*)&v)[0];
    sF[(rp * 2 + 1) * 512 + c] = ((const bf16_t*)&v)[1];
  }
}

// =====================================================================
// Gates r,u via WMMA: [64 rows x 512] @ Wt[64 x 512] per gate.
// waves 0-3: gate r (row tiles 0-3); waves 4-7: gate u.
// Epilogue fuses sigmoid; r-wave writes r*h into X0 h-half; u-wave stores u.
// =====================================================================
__global__ __launch_bounds__(256) void gates_ru_wmma_kernel(
    const bf16_t* __restrict__ Y1f, const bf16_t* __restrict__ Y2f,
    const bf16_t* __restrict__ Y1b, const bf16_t* __restrict__ Y2b,
    const bf16_t* __restrict__ Wrt, const bf16_t* __restrict__ Wut,
    const float* __restrict__ br, const float* __restrict__ bu,
    const float* __restrict__ hst, float* __restrict__ ubuf,
    bf16_t* __restrict__ X0) {
  __shared__ __align__(16) bf16_t sF[64 * 512];
  const int tid = threadIdx.x;
  const int bn0 = blockIdx.x * 64;
  const int b = bn0 >> 10, n0 = bn0 & (NN - 1);
  load_feats_tile(sF, Y1f, Y2f, Y1b, Y2b, b, n0, tid);
  __syncthreads();

  const int lane = tid & 31, wave = tid >> 5;
  const int g = lane >> 4, ml = lane & 15;
  const int mt = wave & 3;
  const int gate = wave >> 2;
  const bf16_t* Wt = gate ? Wut : Wrt;
  v8f acc[4];
  acc[0] = (v8f){}; acc[1] = (v8f){}; acc[2] = (v8f){}; acc[3] = (v8f){};
  const unsigned* frow = (const unsigned*)(sF + (mt * 16 + ml) * 512);

  for (int k0 = 0; k0 < 512; k0 += 32) {
    union { unsigned u[8]; v16bf v; } af;
    const unsigned* ak = frow + (k0 >> 1);
#pragma unroll
    for (int v2 = 0; v2 < 8; ++v2)
      af.u[v2] = ak[g * 8 + v2];
#pragma unroll
    for (int nt = 0; nt < 4; ++nt) {
      v16bf bf = *(const v16bf*)(Wt + (long)(nt * 16 + ml) * 512 + k0 + g * 16);
      acc[nt] = __builtin_amdgcn_wmma_f32_16x16x32_bf16(
          false, af.v, false, bf, (short)0, acc[nt], false, false);
    }
  }

  const float* bias = gate ? bu : br;
#pragma unroll
  for (int nt = 0; nt < 4; ++nt) {
    const int h = nt * 16 + ml;
    const float bv = bias[h];
#pragma unroll
    for (int q = 0; q < 8; ++q) {
      const int row = mt * 16 + g * 8 + q;
      const int idx = (bn0 + row) * 64 + h;
      const float val = 1.f / (1.f + __expf(-(acc[nt][q] + bv)));
      if (gate) ubuf[idx] = val;
      else      X0[((long)(b * 128 + 64 + h)) * NN + (n0 + row)] = (bf16_t)(val * hst[idx]);
    }
  }
}

// =====================================================================
// Candidate gate via WMMA + GRU update: h = u*h + (1-u)*tanh(feats @ Wc + bc)
// 8 waves: mt = wave&3, col-tile half = wave>>2 (2 accumulators each).
// =====================================================================
__global__ __launch_bounds__(256) void gate_c_wmma_kernel(
    const bf16_t* __restrict__ Y1f, const bf16_t* __restrict__ Y2f,
    const bf16_t* __restrict__ Y1b, const bf16_t* __restrict__ Y2b,
    const bf16_t* __restrict__ Wct, const float* __restrict__ bc,
    const float* __restrict__ ubuf, float* __restrict__ hst) {
  __shared__ __align__(16) bf16_t sF[64 * 512];
  const int tid = threadIdx.x;
  const int bn0 = blockIdx.x * 64;
  const int b = bn0 >> 10, n0 = bn0 & (NN - 1);
  load_feats_tile(sF, Y1f, Y2f, Y1b, Y2b, b, n0, tid);
  __syncthreads();

  const int lane = tid & 31, wave = tid >> 5;
  const int g = lane >> 4, ml = lane & 15;
  const int mt = wave & 3;
  const int half = wave >> 2;
  v8f acc[2];
  acc[0] = (v8f){}; acc[1] = (v8f){};
  const unsigned* frow = (const unsigned*)(sF + (mt * 16 + ml) * 512);

  for (int k0 = 0; k0 < 512; k0 += 32) {
    union { unsigned u[8]; v16bf v; } af;
    const unsigned* ak = frow + (k0 >> 1);
#pragma unroll
    for (int v2 = 0; v2 < 8; ++v2)
      af.u[v2] = ak[g * 8 + v2];
#pragma unroll
    for (int nt2 = 0; nt2 < 2; ++nt2) {
      const int nt = half * 2 + nt2;
      v16bf bf = *(const v16bf*)(Wct + (long)(nt * 16 + ml) * 512 + k0 + g * 16);
      acc[nt2] = __builtin_amdgcn_wmma_f32_16x16x32_bf16(
          false, af.v, false, bf, (short)0, acc[nt2], false, false);
    }
  }

#pragma unroll
  for (int nt2 = 0; nt2 < 2; ++nt2) {
    const int h = (half * 2 + nt2) * 16 + ml;
    const float bv = bc[h];
#pragma unroll
    for (int q = 0; q < 8; ++q) {
      const int row = mt * 16 + g * 8 + q;
      const int idx = (bn0 + row) * 64 + h;
      const float cand = tanhf(acc[nt2][q] + bv);
      const float u = ubuf[idx];
      hst[idx] = u * hst[idx] + (1.f - u) * cand;
    }
  }
}

// =====================================================================
// DiffConv sparse part
// =====================================================================
__global__ __launch_bounds__(256) void zero_f32_kernel(float* __restrict__ p, int n) {
  const int i = blockIdx.x * 256 + threadIdx.x;
  if (i < n) p[i] = 0.f;
}

__global__ __launch_bounds__(256) void accum_deg_kernel(const float* __restrict__ ew,
                                                        const int* __restrict__ src,
                                                        const int* __restrict__ dst,
                                                        float* __restrict__ degf,
                                                        float* __restrict__ degb) {
  const int e = blockIdx.x * 256 + threadIdx.x;
  if (e < EE) {
    atomicAdd(&degf[dst[e]], ew[e]);
    atomicAdd(&degb[src[e]], ew[e]);
  }
}

__global__ __launch_bounds__(256) void comp_w_kernel(const float* __restrict__ ew,
                                                     const int* __restrict__ src,
                                                     const int* __restrict__ dst,
                                                     const float* __restrict__ degf,
                                                     const float* __restrict__ degb,
                                                     float* __restrict__ wf,
                                                     float* __restrict__ wb) {
  const int e = blockIdx.x * 256 + threadIdx.x;
  if (e < EE) {
    const float df = degf[dst[e]], db = degb[src[e]];
    wf[e] = (df > 0.f) ? ew[e] / df : 0.f;
    wb[e] = (db > 0.f) ? ew[e] / db : 0.f;
  }
}

__global__ __launch_bounds__(256) void edge_scatter_kernel(const float* __restrict__ in,
                                                           float* __restrict__ out,
                                                           const int* __restrict__ sidx,
                                                           const int* __restrict__ didx,
                                                           const float* __restrict__ w) {
  const int idx = blockIdx.x * 256 + threadIdx.x;     // B*E*H
  const int h = idx & 63;
  const int e = (idx >> 6) & (EE - 1);
  const int b = idx >> 21;
  const float v = in[((long)b * NN + sidx[e]) * HH + h] * w[e];
  atomicAdd(&out[((long)b * NN + didx[e]) * HH + h], v);
}

// =====================================================================
// readout (one-shot, ~0.4 GFLOP): z = feats320 @ W_diff + b_diff; out = z @ W_dec + b_dec
// =====================================================================
__global__ __launch_bounds__(96) void readout_kernel(const float* __restrict__ hst,
                                                     const float* __restrict__ f1,
                                                     const float* __restrict__ f2,
                                                     const float* __restrict__ b1,
                                                     const float* __restrict__ b2,
                                                     const float* __restrict__ Wd,
                                                     const float* __restrict__ bd,
                                                     const float* __restrict__ Wdec,
                                                     const float* __restrict__ bdec,
                                                     float* __restrict__ out) {
  const int bn = blockIdx.x;
  const int b = bn >> 10, n = bn & (NN - 1);
  const int tix = threadIdx.x;
  __shared__ float feat[320];
  __shared__ float z[64];
  const float* Fs[5] = {hst, f1, f2, b1, b2};
  for (int c = tix; c < 320; c += 96) feat[c] = Fs[c / 64][(long)bn * 64 + (c & 63)];
  __syncthreads();
  if (tix < 64) {
    float a = bd[tix];
    for (int c = 0; c < 320; ++c) a += feat[c] * Wd[c * 64 + tix];
    z[tix] = a;
  }
  __syncthreads();
  float a = bdec[tix];
  for (int hh = 0; hh < 64; ++hh) a += z[hh] * Wdec[hh * 96 + tix];
  const int t = tix / 8, f = tix & 7;
  out[((((long)b * HORIZON + t) * NN + n) * FF) + f] = a;
}

// =====================================================================
// host launcher
// =====================================================================
extern "C" void kernel_launch(void* const* d_in, const int* in_sizes, int n_in,
                              void* d_out, int out_size, void* d_ws, size_t ws_size,
                              hipStream_t stream) {
  const float* x    = (const float*)d_in[0];
  const int*   eidx = (const int*)  d_in[1];
  const float* ew   = (const float*)d_in[2];
  const float* adj  = (const float*)d_in[3];
  const float* Wenc = (const float*)d_in[4];
  const float* benc = (const float*)d_in[5];
  const float* nemb = (const float*)d_in[6];
  const float* Wr   = (const float*)d_in[7];
  const float* br   = (const float*)d_in[8];
  const float* Wu   = (const float*)d_in[9];
  const float* bu   = (const float*)d_in[10];
  const float* Wc   = (const float*)d_in[11];
  const float* bc   = (const float*)d_in[12];
  const float* Wd   = (const float*)d_in[13];
  const float* bd   = (const float*)d_in[14];
  const float* Wdec = (const float*)d_in[15];
  const float* bdec = (const float*)d_in[16];
  const int* src = eidx;
  const int* dst = eidx + EE;
  float* out = (float*)d_out;

  char* ws = (char*)d_ws;
  const size_t MB = 1024 * 1024;
  bf16_t* Af  = (bf16_t*)(ws + 0);
  bf16_t* Ab  = (bf16_t*)(ws + 2 * MB);
  bf16_t* X0  = (bf16_t*)(ws + 4 * MB);
  bf16_t* Y1f = (bf16_t*)(ws + 8 * MB);
  bf16_t* Y2f = (bf16_t*)(ws + 12 * MB);
  bf16_t* Y1b = (bf16_t*)(ws + 16 * MB);
  bf16_t* Y2b = (bf16_t*)(ws + 20 * MB);
  float* hst   = (float*)(ws + 24 * MB);   // [B,N,H]
  float* ubuf  = (float*)(ws + 28 * MB);
  float* rsf   = (float*)(ws + 32 * MB);
  float* rsb   = rsf + NN;
  float* degf  = rsb + NN;
  float* degb  = degf + NN;
  float* wf    = degb + NN;
  float* wb    = wf + EE;
  bf16_t* Wrt  = (bf16_t*)(ws + 33 * MB);  // 64KB each
  bf16_t* Wut  = Wrt + 512 * 64;
  bf16_t* Wct  = Wut + 512 * 64;
  float* f1 = (float*)(ws + 34 * MB);
  float* f2 = (float*)(ws + 38 * MB);
  float* b1 = (float*)(ws + 42 * MB);
  float* b2 = (float*)(ws + 46 * MB);

  const int BNH = BB * NN * HH;            // 1,048,576
  const dim3 gemm_grid(COLS / 64, NN / 128);
  const int gate_blocks = (BB * NN) / 64;  // 256

  // --- supports + transposed bf16 gate weights ---
  rowsum_kernel<<<NN, 256, 0, stream>>>(adj, rsf, rsb);
  norm_adj_kernel<<<(NN * NN) / 256, 256, 0, stream>>>(adj, rsf, rsb, Af, Ab);
  transpose_w_kernel<<<128, 256, 0, stream>>>(Wr, Wrt, 512);
  transpose_w_kernel<<<128, 256, 0, stream>>>(Wu, Wut, 512);
  transpose_w_kernel<<<128, 256, 0, stream>>>(Wc, Wct, 512);

  // --- h0 = 0 ---
  zero_f32_kernel<<<BNH / 256, 256, 0, stream>>>(hst, BNH);

  // --- GRU scan over time ---
  for (int t = 0; t < TT; ++t) {
    pack_xh_kernel<<<BNH / 256, 256, 0, stream>>>(x, Wenc, benc, nemb, hst, X0, t);

    gemm_hop_kernel<<<gemm_grid, 256, 0, stream>>>(Af, X0,  Y1f);
    gemm_hop_kernel<<<gemm_grid, 256, 0, stream>>>(Af, Y1f, Y2f);
    gemm_hop_kernel<<<gemm_grid, 256, 0, stream>>>(Ab, X0,  Y1b);
    gemm_hop_kernel<<<gemm_grid, 256, 0, stream>>>(Ab, Y1b, Y2b);

    gates_ru_wmma_kernel<<<gate_blocks, 256, 0, stream>>>(Y1f, Y2f, Y1b, Y2b,
                                                          Wrt, Wut, br, bu,
                                                          hst, ubuf, X0);

    gemm_hop_kernel<<<gemm_grid, 256, 0, stream>>>(Af, X0,  Y1f);
    gemm_hop_kernel<<<gemm_grid, 256, 0, stream>>>(Af, Y1f, Y2f);
    gemm_hop_kernel<<<gemm_grid, 256, 0, stream>>>(Ab, X0,  Y1b);
    gemm_hop_kernel<<<gemm_grid, 256, 0, stream>>>(Ab, Y1b, Y2b);

    gate_c_wmma_kernel<<<gate_blocks, 256, 0, stream>>>(Y1f, Y2f, Y1b, Y2b,
                                                        Wct, bc, ubuf, hst);
  }

  // --- DiffConv sparse readout ---
  zero_f32_kernel<<<(2 * NN + 255) / 256, 256, 0, stream>>>(degf, 2 * NN);
  accum_deg_kernel<<<EE / 256, 256, 0, stream>>>(ew, src, dst, degf, degb);
  comp_w_kernel<<<EE / 256, 256, 0, stream>>>(ew, src, dst, degf, degb, wf, wb);

  const int SCAT = BB * EE * HH;           // 33,554,432
  zero_f32_kernel<<<BNH / 256, 256, 0, stream>>>(f1, BNH);
  edge_scatter_kernel<<<SCAT / 256, 256, 0, stream>>>(hst, f1, src, dst, wf);
  zero_f32_kernel<<<BNH / 256, 256, 0, stream>>>(f2, BNH);
  edge_scatter_kernel<<<SCAT / 256, 256, 0, stream>>>(f1, f2, src, dst, wf);
  zero_f32_kernel<<<BNH / 256, 256, 0, stream>>>(b1, BNH);
  edge_scatter_kernel<<<SCAT / 256, 256, 0, stream>>>(hst, b1, dst, src, wb);
  zero_f32_kernel<<<BNH / 256, 256, 0, stream>>>(b2, BNH);
  edge_scatter_kernel<<<SCAT / 256, 256, 0, stream>>>(b1, b2, dst, src, wb);

  readout_kernel<<<BB * NN, 96, 0, stream>>>(hst, f1, f2, b1, b2, Wd, bd, Wdec, bdec, out);
}
